// LGCN_mlp_9706626089563
// MI455X (gfx1250) — matrix operand order repaired and checked by reading, not compiled, not run on hardware
//
#include <hip/hip_runtime.h>

#define NNODES 50000
#define DIN    128
#define KHOP   8
#define DCAT   1152      // 128 * (8+1)
#define HID    512
#define DOUT   64
#define BN_EPS 1e-5f

typedef __attribute__((ext_vector_type(16))) __bf16        v16bf;
typedef __attribute__((ext_vector_type(16))) unsigned short v16u;
typedef __attribute__((ext_vector_type(8)))  float          v8f;

union BfCast { v16u u; v16bf b; };

// round-to-nearest-even f32 -> bf16 (bit arithmetic; no fp-convert needed)
static __device__ __forceinline__ unsigned short f2bf(float f) {
  unsigned u = __float_as_uint(f);
  u += 0x7FFFu + ((u >> 16) & 1u);
  return (unsigned short)(u >> 16);
}

static __device__ __forceinline__ v16bf pack16(float4 a0, float4 a1, float4 a2, float4 a3) {
  BfCast A;
  A.u[0] = f2bf(a0.x); A.u[1] = f2bf(a0.y); A.u[2]  = f2bf(a0.z); A.u[3]  = f2bf(a0.w);
  A.u[4] = f2bf(a1.x); A.u[5] = f2bf(a1.y); A.u[6]  = f2bf(a1.z); A.u[7]  = f2bf(a1.w);
  A.u[8] = f2bf(a2.x); A.u[9] = f2bf(a2.y); A.u[10] = f2bf(a2.z); A.u[11] = f2bf(a2.w);
  A.u[12] = f2bf(a3.x); A.u[13] = f2bf(a3.y); A.u[14] = f2bf(a3.z); A.u[15] = f2bf(a3.w);
  return A.b;
}

static __device__ __forceinline__ v8f wmma_bf16(v16bf a, v16bf b, v8f c) {
  // 8 args: (neg_a, A, neg_b, B, c_mod, C, reuse_a, reuse_b)
  return __builtin_amdgcn_wmma_f32_16x16x32_bf16(false, a, false, b, (short)0, c, false, false);
}

// ---------------- graph-normalization kernels ----------------

__global__ void k_init(float* deg, float* sumb, float* sqb) {
  int i = blockIdx.x * blockDim.x + threadIdx.x;
  if (i < NNODES) deg[i] = 1.0f;            // self-loop contribution to in-degree
  if (i < HID) { sumb[i] = 0.f; sqb[i] = 0.f; }
}

__global__ void k_deg(const long long* __restrict__ ei, int E, float* __restrict__ deg) {
  int e = blockIdx.x * blockDim.x + threadIdx.x;
  if (e >= E) return;
  long long c = ei[(size_t)E + e];          // col (target)
  atomicAdd(&deg[c], 1.0f);
}

__global__ void k_dinv(const float* __restrict__ deg, float* __restrict__ dinv) {
  int i = blockIdx.x * blockDim.x + threadIdx.x;
  if (i >= NNODES) return;
  float d = deg[i];
  dinv[i] = d > 0.f ? rsqrtf(d) : 0.f;
}

__global__ void k_norm(const long long* __restrict__ ei, int E,
                       const float* __restrict__ dinv, float* __restrict__ norm) {
  int idx = blockIdx.x * blockDim.x + threadIdx.x;
  if (idx >= E + NNODES) return;
  if (idx < E) {
    norm[idx] = dinv[ei[idx]] * dinv[ei[(size_t)E + idx]];
  } else {
    float d = dinv[idx - E];
    norm[idx] = d * d;
  }
}

// ---------------- K-hop propagation ----------------

__global__ void k_copy(const float* __restrict__ feat, float* __restrict__ xcat) {
  int idx = blockIdx.x * blockDim.x + threadIdx.x;   // N*128 threads
  if (idx >= NNODES * DIN) return;
  int i = idx >> 7, c = idx & 127;
  xcat[(size_t)i * DCAT + c] = feat[idx];
}

// x_next[i] = dinv[i]^2 * x_prev[i]   (self-loop term; full overwrite => no pre-zero)
__global__ void k_selfloop(const float* __restrict__ dinv, const float* __restrict__ xcat,
                           int off_prev, int off_next, float* __restrict__ xcat_w) {
  int idx = blockIdx.x * blockDim.x + threadIdx.x;
  if (idx >= NNODES * DIN) return;
  int i = idx >> 7, c = idx & 127;
  float d = dinv[i];
  xcat_w[(size_t)i * DCAT + off_next + c] = d * d * xcat[(size_t)i * DCAT + off_prev + c];
}

// one 32-lane wave slice per edge: 32 lanes x float4 = 128 features
__global__ void k_scatter(const long long* __restrict__ ei, int E,
                          const float* __restrict__ norm,
                          const float* __restrict__ xprev,
                          float* __restrict__ xnext) {
  long long tid = (long long)blockIdx.x * blockDim.x + threadIdx.x;
  int e = (int)(tid >> 5);
  if (e >= E) return;
  int c = ((int)tid & 31) * 4;
  long long r = ei[e];
  long long t = ei[(size_t)E + e];
  float w = norm[e];
  const float4 v = *(const float4*)(xprev + (size_t)r * DCAT + c);
  float* dst = xnext + (size_t)t * DCAT + c;
  atomicAdd(dst + 0, w * v.x);
  atomicAdd(dst + 1, w * v.y);
  atomicAdd(dst + 2, w * v.z);
  atomicAdd(dst + 3, w * v.w);
}

// ---------------- weight transpose + bf16 convert ----------------
// Wt[n*K + k] = bf16(W[k*N + n])  -> B-operand loads become 32B contiguous
__global__ void k_tconv(const float* __restrict__ W, unsigned short* __restrict__ Wt,
                        int K, int Nn) {
  int idx = blockIdx.x * blockDim.x + threadIdx.x;
  if (idx >= K * Nn) return;
  int n = idx / K, k = idx - n * K;
  Wt[idx] = f2bf(W[(size_t)k * Nn + n]);
}

// ---------------- GEMM1: h = leaky_relu(x_cat @ W1 + b1), bf16 WMMA ----------------
// wave = one 16(M) x 64(N) strip; 3125 m-tiles * 8 n-groups = 25000 waves
__global__ void __launch_bounds__(256) k_gemm1(const float* __restrict__ xcat,
                                               const unsigned short* __restrict__ w1t,
                                               const float* __restrict__ b1,
                                               float* __restrict__ h) {
  int wave = blockIdx.x * 8 + (threadIdx.x >> 5);
  int lane = threadIdx.x & 31;
  int l16 = lane & 15, hi = lane >> 4;
  int mt = wave >> 3;        // 0..3124
  int nb = (wave & 7) * 64;  // n-group base

  const float* arow = xcat + (size_t)(mt * 16 + l16) * DCAT + hi * 16;
  v8f acc[4] = {};

  for (int k0 = 0; k0 < DCAT; k0 += 32) {
    const float* ap = arow + k0;
    __builtin_prefetch(ap + 64, 0, 0);
    float4 a0 = *(const float4*)(ap);
    float4 a1 = *(const float4*)(ap + 4);
    float4 a2 = *(const float4*)(ap + 8);
    float4 a3 = *(const float4*)(ap + 12);
    v16bf a = pack16(a0, a1, a2, a3);
    const unsigned short* bbase = w1t + k0 + hi * 16;
#pragma unroll
    for (int nn = 0; nn < 4; nn++) {
      v16bf b = *(const v16bf*)(bbase + (size_t)(nb + nn * 16 + l16) * DCAT);
      acc[nn] = wmma_bf16(a, b, acc[nn]);
    }
  }

#pragma unroll
  for (int nn = 0; nn < 4; nn++) {
    int ncol = nb + nn * 16 + l16;
    float bias = b1[ncol];
#pragma unroll
    for (int v = 0; v < 8; v++) {
      int mr = mt * 16 + v + hi * 8;
      float val = acc[nn][v] + bias;
      h[(size_t)mr * HID + ncol] = val > 0.f ? val : 0.2f * val;
    }
  }
}

// ---------------- BatchNorm statistics ----------------

__global__ void k_bnstat(const float* __restrict__ h, float* __restrict__ sumb,
                         float* __restrict__ sqb) {
  int r0 = blockIdx.x * 64;
  int c = threadIdx.x;  // 0..255 ; covers cols c and c+256
  float s0 = 0.f, q0 = 0.f, s1 = 0.f, q1 = 0.f;
  int rmax = NNODES - r0; if (rmax > 64) rmax = 64;
  for (int r = 0; r < rmax; r++) {
    const float* hp = h + (size_t)(r0 + r) * HID;
    float a = hp[c], b = hp[c + 256];
    s0 += a; q0 += a * a; s1 += b; q1 += b * b;
  }
  atomicAdd(&sumb[c], s0);        atomicAdd(&sqb[c], q0);
  atomicAdd(&sumb[c + 256], s1);  atomicAdd(&sqb[c + 256], q1);
}

__global__ void k_bnfin(const float* __restrict__ sumb, const float* __restrict__ sqb,
                        const float* __restrict__ gamma, const float* __restrict__ beta,
                        float* __restrict__ scale, float* __restrict__ shift) {
  int c = blockIdx.x * blockDim.x + threadIdx.x;
  if (c >= HID) return;
  float inv_n = 1.0f / (float)NNODES;
  float mean = sumb[c] * inv_n;
  float var  = sqb[c] * inv_n - mean * mean;
  float rstd = rsqrtf(var + BN_EPS);
  float sc = gamma[c] * rstd;
  scale[c] = sc;
  shift[c] = beta[c] - mean * sc;
}

// ---------------- GEMM2: out = ((h*scale+shift) @ W2) + b2, bf16 WMMA ----------------
// wave = 16(M) x 64(N=all); BN affine fused into the A-side bf16 convert
__global__ void __launch_bounds__(256) k_gemm2(const float* __restrict__ h,
                                               const unsigned short* __restrict__ w2t,
                                               const float* __restrict__ scale,
                                               const float* __restrict__ shift,
                                               const float* __restrict__ b2,
                                               float* __restrict__ out) {
  int wave = blockIdx.x * 8 + (threadIdx.x >> 5);
  if (wave >= NNODES / 16) return;
  int lane = threadIdx.x & 31;
  int l16 = lane & 15, hi = lane >> 4;

  const float* arow = h + (size_t)(wave * 16 + l16) * HID + hi * 16;
  v8f acc[4] = {};

  for (int k0 = 0; k0 < HID; k0 += 32) {
    const float* ap = arow + k0;
    const float* sp = scale + k0 + hi * 16;
    const float* tp = shift + k0 + hi * 16;
    float4 a0 = *(const float4*)(ap);       float4 a1 = *(const float4*)(ap + 4);
    float4 a2 = *(const float4*)(ap + 8);   float4 a3 = *(const float4*)(ap + 12);
    float4 s0 = *(const float4*)(sp);       float4 s1 = *(const float4*)(sp + 4);
    float4 s2 = *(const float4*)(sp + 8);   float4 s3 = *(const float4*)(sp + 12);
    float4 t0 = *(const float4*)(tp);       float4 t1 = *(const float4*)(tp + 4);
    float4 t2 = *(const float4*)(tp + 8);   float4 t3 = *(const float4*)(tp + 12);
    float4 f0 = {fmaf(a0.x, s0.x, t0.x), fmaf(a0.y, s0.y, t0.y), fmaf(a0.z, s0.z, t0.z), fmaf(a0.w, s0.w, t0.w)};
    float4 f1 = {fmaf(a1.x, s1.x, t1.x), fmaf(a1.y, s1.y, t1.y), fmaf(a1.z, s1.z, t1.z), fmaf(a1.w, s1.w, t1.w)};
    float4 f2 = {fmaf(a2.x, s2.x, t2.x), fmaf(a2.y, s2.y, t2.y), fmaf(a2.z, s2.z, t2.z), fmaf(a2.w, s2.w, t2.w)};
    float4 f3 = {fmaf(a3.x, s3.x, t3.x), fmaf(a3.y, s3.y, t3.y), fmaf(a3.z, s3.z, t3.z), fmaf(a3.w, s3.w, t3.w)};
    v16bf a = pack16(f0, f1, f2, f3);
    const unsigned short* bbase = w2t + k0 + hi * 16;
#pragma unroll
    for (int nn = 0; nn < 4; nn++) {
      v16bf b = *(const v16bf*)(bbase + (size_t)(nn * 16 + l16) * HID);
      acc[nn] = wmma_bf16(a, b, acc[nn]);
    }
  }

#pragma unroll
  for (int nn = 0; nn < 4; nn++) {
    int ncol = nn * 16 + l16;
    float bias = b2[ncol];
#pragma unroll
    for (int v = 0; v < 8; v++) {
      int mr = wave * 16 + v + hi * 8;
      out[(size_t)mr * DOUT + ncol] = acc[nn][v] + bias;
    }
  }
}

// ---------------- host orchestration ----------------

extern "C" void kernel_launch(void* const* d_in, const int* in_sizes, int n_in,
                              void* d_out, int out_size, void* d_ws, size_t ws_size,
                              hipStream_t stream) {
  const float*     feature = (const float*)d_in[0];
  const long long* ei      = (const long long*)d_in[1];   // int64 [2, E]
  const float*     W1      = (const float*)d_in[2];
  const float*     b1      = (const float*)d_in[3];
  const float*     gamma   = (const float*)d_in[4];
  const float*     beta    = (const float*)d_in[5];
  const float*     W2      = (const float*)d_in[6];
  const float*     b2      = (const float*)d_in[7];
  const int E = in_sizes[1] / 2;

  float* out  = (float*)d_out;                    // [N x 64]
  float* norm = out + (size_t)NNODES * DOUT;      // [E + N]

  // workspace carve-out (256B aligned chunks); needs ~335 MB
  char* p = (char*)d_ws;
  auto carve = [&](size_t bytes) -> char* {
    char* r = p; p += (bytes + 255) & ~(size_t)255; return r;
  };
  float*          deg   = (float*)carve((size_t)NNODES * 4);
  float*          dinv  = (float*)carve((size_t)NNODES * 4);
  float*          sumb  = (float*)carve(HID * 4);
  float*          sqb   = (float*)carve(HID * 4);
  float*          scale = (float*)carve(HID * 4);
  float*          shift = (float*)carve(HID * 4);
  unsigned short* w1t   = (unsigned short*)carve((size_t)DCAT * HID * 2);
  unsigned short* w2t   = (unsigned short*)carve((size_t)HID * DOUT * 2);
  float*          xcat  = (float*)carve((size_t)NNODES * DCAT * 4);
  float*          h     = (float*)carve((size_t)NNODES * HID * 4);

  const int B = 256;
  // graph norms
  k_init<<<(NNODES + B - 1) / B, B, 0, stream>>>(deg, sumb, sqb);
  k_deg<<<(E + B - 1) / B, B, 0, stream>>>(ei, E, deg);
  k_dinv<<<(NNODES + B - 1) / B, B, 0, stream>>>(deg, dinv);
  k_norm<<<(E + NNODES + B - 1) / B, B, 0, stream>>>(ei, E, dinv, norm);
  // weights -> transposed bf16
  k_tconv<<<((DCAT * HID) + B - 1) / B, B, 0, stream>>>(W1, w1t, DCAT, HID);
  k_tconv<<<((HID * DOUT) + B - 1) / B, B, 0, stream>>>(W2, w2t, HID, DOUT);
  // hop 0
  k_copy<<<(NNODES * DIN + B - 1) / B, B, 0, stream>>>(feature, xcat);
  // hops 1..K
  for (int k = 1; k <= KHOP; k++) {
    int op = (k - 1) * DIN, on = k * DIN;
    k_selfloop<<<(NNODES * DIN + B - 1) / B, B, 0, stream>>>(dinv, xcat, op, on, xcat);
    long long thr = (long long)E * 32;
    k_scatter<<<(unsigned)((thr + B - 1) / B), B, 0, stream>>>(ei, E, norm,
                                                              xcat + op, xcat + on);
  }
  // dense path
  k_gemm1<<<(NNODES / 16) * 8 / 8, B, 0, stream>>>(xcat, w1t, b1, h);   // 3125 blocks
  k_bnstat<<<(NNODES + 63) / 64, B, 0, stream>>>(h, sumb, sqb);
  k_bnfin<<<(HID + B - 1) / B, B, 0, stream>>>(sumb, sqb, gamma, beta, scale, shift);
  k_gemm2<<<((NNODES / 16) + 7) / 8, B, 0, stream>>>(h, w2t, scale, shift, b2, out);
}
